// IrregularClinicalNotesModel_44487271252164
// MI455X (gfx1250) — matrix operand order repaired and checked by reading, not compiled, not run on hardware
//
#include <hip/hip_runtime.h>
#include <hip/hip_bf16.h>

// ---------------------------------------------------------------------------
// Types for CDNA5 WMMA (wave32): A/B = 16 x bf16 per lane, C/D = 8 x f32.
// ---------------------------------------------------------------------------
typedef __bf16 bf16;
typedef __attribute__((ext_vector_type(16))) __bf16 bf16x16;
typedef __attribute__((ext_vector_type(8)))  float  f32x8;

static __device__ __forceinline__ f32x8 wmma_bf16(bf16x16 a, bf16x16 b, f32x8 c) {
  // 8 args: (neg_a, A, neg_b, B, c_mod, C, reuse_a, reuse_b)
  return __builtin_amdgcn_wmma_f32_16x16x32_bf16(false, a, false, b, (short)0, c,
                                                 false, false);
}

// ---------------------------------------------------------------------------
// CDNA5 async global->LDS copy (ASYNCcnt path, ISA 10.7 / 15.18.3).
// Each issue moves 16 bytes per lane straight into LDS (no VGPR round trip).
// ---------------------------------------------------------------------------
static __device__ __forceinline__ void async_b128(unsigned ldsoff, const void* gaddr) {
  asm volatile("global_load_async_to_lds_b128 %0, %1, off"
               :: "v"(ldsoff), "v"(gaddr) : "memory");
}
static __device__ __forceinline__ void async_wait0() {
  asm volatile("s_wait_asynccnt 0x0" ::: "memory");
}
static __device__ __forceinline__ unsigned lds_off(const void* p) {
  // generic pointers to LDS carry the LDS byte offset in the low 32 bits
  return (unsigned)(size_t)p;
}

// ---------------------------------------------------------------------------
// Fragment helpers (A/B: this file's contiguous k order, not the ISA table; C/D per ISA 7.12.2)
// A 16x32 bf16: lane m = lane&15; kbase = (lane<16)?0:16; elems 0..7 -> k=kb..kb+7,
//               elems 8..15 -> k=kb+8..kb+15  (contiguous K run of 16)
// B 32x16 bf16: lane n = lane&15; kbase = (lane<16)?0:16; elems i -> k=kb+i
// C/D 16x16 f32: lane n = lane&15; VGPR r -> m = r + ((lane<16)?0:8)
// ---------------------------------------------------------------------------
static __device__ __forceinline__ bf16x16
afrag_f32(const float* base, int ld, int lane, int k0, int kmax) {
  int m  = lane & 15;
  int kb = k0 + ((lane & 16) ? 16 : 0);
  const float* p = base + m * ld;
  bf16x16 r;
#pragma unroll
  for (int i = 0; i < 8; ++i) {
    int ka = kb + i, kc = kb + 8 + i;
    r[i]     = (ka < kmax) ? (bf16)p[ka] : (bf16)0.0f;
    r[i + 8] = (kc < kmax) ? (bf16)p[kc] : (bf16)0.0f;
  }
  return r;
}

static __device__ __forceinline__ bf16x16
afrag_b16(const bf16* base, int ld, int lane, int k0, int kmax) {
  int m  = lane & 15;
  int kb = k0 + ((lane & 16) ? 16 : 0);
  const bf16* p = base + m * ld;
  bf16x16 r;
#pragma unroll
  for (int i = 0; i < 8; ++i) {
    int ka = kb + i, kc = kb + 8 + i;
    r[i]     = (ka < kmax) ? p[ka] : (bf16)0.0f;
    r[i + 8] = (kc < kmax) ? p[kc] : (bf16)0.0f;
  }
  return r;
}

// B fragment from LDS tile stored transposed: bt[n][ld] holding K contiguous.
static __device__ __forceinline__ bf16x16
bfrag_bt(const bf16* bt, int ld, int lane, int k0) {
  int n  = lane & 15;
  int kb = k0 + ((lane & 16) ? 16 : 0);
  const bf16* p = bt + n * ld + kb;
  bf16x16 r;
#pragma unroll
  for (int i = 0; i < 16; ++i) r[i] = p[i];
  return r;
}

// B fragment where logical B[k][n] = S[n*ld + k] (transpose-on-read, bf16 src).
static __device__ __forceinline__ bf16x16
bfrag_rowT(const bf16* S, int ld, int lane, int k0, int kmax) {
  int n  = lane & 15;
  int kb = (lane & 16) ? 16 : 0;
  const bf16* p = S + n * ld;
  bf16x16 r;
#pragma unroll
  for (int i = 0; i < 16; ++i) {
    int kk = k0 + kb + i;
    r[i] = (kk < kmax) ? p[kk] : (bf16)0.0f;
  }
  return r;
}

static __device__ __forceinline__ void
cstore_f32_scale(float* dst, int ld, int lane, f32x8 c, float s) {
  int n = lane & 15, mb = (lane & 16) ? 8 : 0;
#pragma unroll
  for (int r = 0; r < 8; ++r) dst[(mb + r) * ld + n] = c[r] * s;
}

static __device__ __forceinline__ void
cstore_f32_add(float* dst, int ld, int lane, f32x8 c) {
  int n = lane & 15, mb = (lane & 16) ? 8 : 0;
#pragma unroll
  for (int r = 0; r < 8; ++r) dst[(mb + r) * ld + n] += c[r];
}

static __device__ __forceinline__ void
cstore_b16(bf16* dst, int ld, int lane, f32x8 c) {
  int n = lane & 15, mb = (lane & 16) ? 8 : 0;
#pragma unroll
  for (int r = 0; r < 8; ++r) dst[(mb + r) * ld + n] = (bf16)c[r];
}

// Transposed store: dst[n*ldT + moff + m]
static __device__ __forceinline__ void
cstore_b16T(bf16* dst, int ldT, int moff, int lane, f32x8 c) {
  int n = lane & 15, mb = (lane & 16) ? 8 : 0;
#pragma unroll
  for (int r = 0; r < 8; ++r) dst[n * ldT + moff + mb + r] = (bf16)c[r];
}

// ---------------------------------------------------------------------------
// Async-stage a 128-row x 32-K bf16 B-tile (pre-transposed weights W^T[N][K])
// into bt[128][40] (KP=40). 256 threads, 2 x 16B chunks each.
// ---------------------------------------------------------------------------
static __device__ __forceinline__ void
stage_bt_async(bf16* bt, const bf16* WT, int ldk, int k0, int n0, int tid) {
#pragma unroll
  for (int j = 0; j < 2; ++j) {
    int cid = tid + (j << 8);
    int n = cid >> 2, c = (cid & 3) << 4;
    unsigned loff = lds_off((const char*)bt + n * 80 + c);
    const char* g = (const char*)(WT + (size_t)(n0 + n) * ldk + k0) + c;
    async_b128(loff, g);
  }
}

// ---------------------------------------------------------------------------
// Model dimensions
// ---------------------------------------------------------------------------
#define Bsz   256
#define Lsz   64
#define D_T   768
#define DH    128
#define ALPHA 48
#define DTE   64
#define Hn    8
#define NLl   3
#define DFF   512

// ---------------------------------------------------------------------------
// Double-buffered 48xN GEMM: acc[3] += A(48 x Ktot) @ W^T-tile, wave w owns
// N-tile w.  B staged via async copies, next tile issued while computing.
// ---------------------------------------------------------------------------
template <typename AF>
static __device__ __forceinline__ void
gemm48_db(f32x8* acc, const bf16* WT, int ldk, int n0, int Ktot,
          bf16* bt0, bf16* bt1, int tid, int lane, int w, AF&& af) {
  stage_bt_async(bt0, WT, ldk, 0, n0, tid);
  int nk = Ktot >> 5;
  for (int i = 0; i < nk; ++i) {
    bf16* cur = (i & 1) ? bt1 : bt0;
    bf16* nxt = (i & 1) ? bt0 : bt1;
    async_wait0();
    __syncthreads();
    if (i + 1 < nk) stage_bt_async(nxt, WT, ldk, (i + 1) << 5, n0, tid);
    int k0 = i << 5;
    bf16x16 bfv = bfrag_bt(cur + (w * 16) * 40, 40, lane, 0);
    bf16x16 a0 = af(0, k0), a1 = af(1, k0), a2 = af(2, k0);
    acc[0] = wmma_bf16(a0, bfv, acc[0]);
    acc[1] = wmma_bf16(a1, bfv, acc[1]);
    acc[2] = wmma_bf16(a2, bfv, acc[2]);
  }
  __syncthreads();
}

// ---------------------------------------------------------------------------
// K0: fp32 -> bf16 weight conversion WITH transpose: dst[l][n][k] = src[l][k][n]
// ---------------------------------------------------------------------------
__global__ void cvtT_f32_bf16(const float* __restrict__ s, bf16* __restrict__ d,
                              int K, int N, int total) {
  int i = blockIdx.x * 256 + threadIdx.x;
  if (i >= total) return;
  int kn = K * N;
  int l = i / kn, rem = i - l * kn;
  int k = rem / N, n = rem - k * N;
  d[l * kn + n * K + k] = (bf16)s[i];
}

// ---------------------------------------------------------------------------
// K0b: precompute q = time_embed(0..47) @ wq  ->  qvec[48][64] fp32
// ---------------------------------------------------------------------------
__global__ void qprep_kernel(const float* __restrict__ te_w, const float* __restrict__ te_b,
                             const float* __restrict__ wq, float* __restrict__ qvec) {
  int a = threadIdx.x;
  if (a >= ALPHA) return;
  float t = (float)a;
  float te[DTE];
#pragma unroll
  for (int j = 0; j < DTE; ++j) {
    float lin = t * te_w[j] + te_b[j];
    te[j] = (j == 0) ? lin : __sinf(lin);
  }
  for (int d = 0; d < DTE; ++d) {
    float s = 0.f;
#pragma unroll
    for (int j = 0; j < DTE; ++j) s += te[j] * wq[j * DTE + d];
    qvec[a * DTE + d] = s;
  }
}

// ---------------------------------------------------------------------------
// K1: note_emb^T = (cls_emb[16384x768] @ proj_w + proj_b) stored per-batch
//     transposed:  outT[b][f(128)][l(64)]  (bf16).
//     256 blocks x 256 thr; block tile 64x128; double-buffered async staging
//     of both A (fp32) and B (bf16 W^T) tiles.  This is the only op near the
//     23.3 TB/s HBM roofline (50 MB of cls_emb).
// ---------------------------------------------------------------------------
__global__ void __launch_bounds__(256)
proj_gemm_kernel(const float* __restrict__ A, const bf16* __restrict__ WT,
                 const float* __restrict__ bias, bf16* __restrict__ outT) {
  extern __shared__ unsigned char sm[];
  float* Asl0 = (float*)sm;            // [64][40] f32 = 10240 B
  float* Asl1 = (float*)(sm + 10240);  // [64][40] f32 = 10240 B
  bf16*  Btl0 = (bf16*)(sm + 20480);   // [128][40] bf16 = 10240 B
  bf16*  Btl1 = (bf16*)(sm + 30720);   // [128][40] bf16 = 10240 B
  // total dynamic LDS = 40960 bytes

  int tid = threadIdx.x, lane = tid & 31, w = tid >> 5;
  int mt = w & 3, ng = w >> 2;        // wave -> (m-tile, group of 4 n-tiles)
  int m0 = blockIdx.x * 64;

  const f32x8 vz = {};
  f32x8 acc[4] = {vz, vz, vz, vz};

  auto stageA = [&](float* buf, int k0) {
#pragma unroll
    for (int j = 0; j < 2; ++j) {
      int cid = tid + (j << 8);
      int row = cid >> 3, c = (cid & 7) << 4;
      unsigned loff = lds_off((const char*)buf + row * 160 + c);
      const char* g = (const char*)(A + (size_t)(m0 + row) * D_T + k0) + c;
      async_b128(loff, g);
    }
  };

  stageA(Asl0, 0);
  stage_bt_async(Btl0, WT, D_T, 0, 0, tid);

  for (int i = 0; i < D_T / 32; ++i) {
    float* Ac = (i & 1) ? Asl1 : Asl0;
    bf16*  Bc = (i & 1) ? Btl1 : Btl0;
    float* An = (i & 1) ? Asl0 : Asl1;
    bf16*  Bn = (i & 1) ? Btl0 : Btl1;
    async_wait0();
    __syncthreads();
    if (i + 1 < D_T / 32) {
      stageA(An, (i + 1) * 32);
      stage_bt_async(Bn, WT, D_T, (i + 1) * 32, 0, tid);
    }
    bf16x16 af = afrag_f32(Ac + (mt * 16) * 40, 40, lane, 0, 1 << 30);
    bf16x16 bv[4];
#pragma unroll
    for (int j = 0; j < 4; ++j)
      bv[j] = bfrag_bt(Bc + ((ng * 4 + j) * 16) * 40, 40, lane, 0);
#pragma unroll
    for (int j = 0; j < 4; ++j) acc[j] = wmma_bf16(af, bv[j], acc[j]);
    __syncthreads();
  }

  int n = lane & 15, mb = (lane & 16) ? 8 : 0;
#pragma unroll
  for (int j = 0; j < 4; ++j) {
    int col = (ng * 4 + j) * 16 + n;
    float bv = bias[col];
#pragma unroll
    for (int r = 0; r < 8; ++r) {
      int row = m0 + mt * 16 + mb + r;     // global note index
      int b = row >> 6, l = row & 63;
      outT[((size_t)b << 13) + col * Lsz + l] = (bf16)(acc[j][r] + bv);
    }
  }
}

// ---------------------------------------------------------------------------
// K2: per-batch time-attention pooling.
//     k_te = time_embed(note_times[b]) @ wk ; scores = q.k/sqrt(8) ; mask ;
//     softmax over L ; abar = mean_h attn ; x[b] = abar(48x64) @ note_emb[b]
//     note_emb^T tile is async-copied early, overlapping the VALU phase.
// ---------------------------------------------------------------------------
__global__ void __launch_bounds__(256)
pool_kernel(const float* __restrict__ note_times, const int* __restrict__ note_counts,
            const float* __restrict__ te_w, const float* __restrict__ te_b,
            const float* __restrict__ wk, const float* __restrict__ qvec,
            const bf16* __restrict__ nembT, float* __restrict__ xout) {
  __shared__ float klds[Lsz * DTE];     // [l][d]
  __shared__ float abar[ALPHA * Lsz];   // [a][l]
  __shared__ bf16  bt[DH * 72];         // note_emb^T [f][l], KP=72

  int b = blockIdx.x, tid = threadIdx.x, lane = tid & 31, w = tid >> 5;

  // issue async copy of note_emb^T[b] (128 x 64 bf16) NOW; consume after softmax
  {
    const char* src = (const char*)(nembT + ((size_t)b << 13));
#pragma unroll
    for (int j = 0; j < 4; ++j) {
      int cid = tid + (j << 8);
      int f = cid >> 3, c = (cid & 7) << 4;
      async_b128(lds_off((const char*)bt + f * 144 + c), src + f * 128 + c);
    }
  }

  for (int i = tid; i < ALPHA * Lsz; i += 256) abar[i] = 0.f;

  if (tid < Lsz) {
    float t = note_times[b * Lsz + tid];
    float te[DTE];
#pragma unroll
    for (int j = 0; j < DTE; ++j) {
      float lin = t * te_w[j] + te_b[j];
      te[j] = (j == 0) ? lin : __sinf(lin);
    }
    for (int d = 0; d < DTE; ++d) {
      float a = 0.f;
#pragma unroll
      for (int j = 0; j < DTE; ++j) a += te[j] * wk[j * DTE + d];
      klds[tid * DTE + d] = a;
    }
  }
  __syncthreads();

  int cnt = note_counts[b];
  for (int row = tid; row < Hn * ALPHA; row += 256) {
    int h = row / ALPHA, a = row % ALPHA;
    float q8[8];
#pragma unroll
    for (int j = 0; j < 8; ++j) q8[j] = qvec[a * DTE + h * 8 + j];
    float sarr[Lsz];
    float mx = -3.4e38f;
#pragma unroll
    for (int l = 0; l < Lsz; ++l) {
      float s;
      if (l < cnt) {
        float d = 0.f;
#pragma unroll
        for (int j = 0; j < 8; ++j) d += q8[j] * klds[l * DTE + h * 8 + j];
        s = d * 0.35355339059327373f;  // 1/sqrt(8)
      } else {
        s = -1.0e9f;
      }
      sarr[l] = s;
      mx = fmaxf(mx, s);
    }
    float sum = 0.f;
#pragma unroll
    for (int l = 0; l < Lsz; ++l) { float e = __expf(sarr[l] - mx); sarr[l] = e; sum += e; }
    float inv = 0.125f / sum;  // includes mean over H=8
#pragma unroll
    for (int l = 0; l < Lsz; ++l) atomicAdd(&abar[a * Lsz + l], sarr[l] * inv);
  }
  async_wait0();
  __syncthreads();

  const f32x8 vz = {};
  f32x8 acc[3] = {vz, vz, vz};
  for (int k0 = 0; k0 < Lsz; k0 += 32) {
    bf16x16 bfv = bfrag_bt(bt + (w * 16) * 72, 72, lane, k0);
#pragma unroll
    for (int mtt = 0; mtt < 3; ++mtt) {
      bf16x16 af = afrag_f32(abar + mtt * 16 * Lsz, Lsz, lane, k0, 1 << 30);
      acc[mtt] = wmma_bf16(af, bfv, acc[mtt]);
    }
  }

  float mask = (cnt > 0) ? 1.f : 0.f;
  float* xb = xout + (size_t)b * ALPHA * DH;
  int n = lane & 15, mb = (lane & 16) ? 8 : 0;
#pragma unroll
  for (int mtt = 0; mtt < 3; ++mtt) {
    int col = w * 16 + n;
#pragma unroll
    for (int r = 0; r < 8; ++r)
      xb[(mtt * 16 + mb + r) * DH + col] = acc[mtt][r] * mask;
  }
}

// ---------------------------------------------------------------------------
// Layer-norm over the 128 features of each of 48 rows (thread-per-row)
// ---------------------------------------------------------------------------
static __device__ __forceinline__ void
ln_rows(float* x, const float* __restrict__ g, const float* __restrict__ bb, int tid) {
  if (tid < ALPHA) {
    float* p = x + tid * DH;
    float s = 0.f, ss = 0.f;
#pragma unroll 8
    for (int f = 0; f < DH; ++f) { float v = p[f]; s += v; ss += v * v; }
    float mu  = s * (1.f / DH);
    float var = ss * (1.f / DH) - mu * mu;
    float inv = rsqrtf(var + 1e-5f);
#pragma unroll 8
    for (int f = 0; f < DH; ++f) p[f] = (p[f] - mu) * inv * g[f] + bb[f];
  }
}

// ---------------------------------------------------------------------------
// K3: fused 3-layer transformer (seq=48, DH=128, H=8, hd=16, DFF=512)
//     + classifier head.  One workgroup (8 waves) per batch element; x lives
//     in LDS for the whole kernel; weights (W^T, bf16, L2-resident) are
//     double-buffered via async copies.  wave w == head h == N-tile w.
// ---------------------------------------------------------------------------
__global__ void __launch_bounds__(256)
tfm_kernel(float* __restrict__ xglob,
           const bf16* __restrict__ wqt, const bf16* __restrict__ wkt,
           const bf16* __restrict__ wvt, const bf16* __restrict__ wot,
           const bf16* __restrict__ w1t, const bf16* __restrict__ w2t,
           const float* __restrict__ ln1g, const float* __restrict__ ln1b,
           const float* __restrict__ ln2g, const float* __restrict__ ln2b,
           const float* __restrict__ fb1, const float* __restrict__ fb2,
           const float* __restrict__ cw1, const float* __restrict__ cb1,
           const float* __restrict__ cw2, const float* __restrict__ cb2,
           float* __restrict__ out) {
  extern __shared__ unsigned char smem[];
  float* x    = (float*)smem;                         // 48*128*4  = 24576
  bf16* qb    = (bf16*)(smem + 24576);                // 48*128*2  = 12288
  bf16* kbq   = (bf16*)(smem + 36864);                // 12288
  bf16* vbT   = (bf16*)(smem + 49152);                // [f][a] 128*48*2 = 12288
  unsigned char* scr = smem + 61440;                  // 86016 scratch
  float* scores = (float*)scr;                        // [h][48][48] f32 = 73728
  bf16*  o1b    = (bf16*)(scr + 73728);               // [48][128] = 12288
  bf16*  h1b    = (bf16*)scr;                         // [48][512] = 49152 (aliases scores)
  bf16*  bt0    = (bf16*)(smem + 147456);             // [128][40] = 10240
  bf16*  bt1    = (bf16*)(smem + 157696);             // [128][40] = 10240
  // total dynamic LDS = 167936 bytes

  int b = blockIdx.x, tid = threadIdx.x, lane = tid & 31, w = tid >> 5;
  const f32x8 vz = {};

  auto af_x  = [&](int mt, int k0) { return afrag_f32(x + mt * 16 * DH, DH, lane, k0, 1 << 30); };
  auto af_o1 = [&](int mt, int k0) { return afrag_b16(o1b + mt * 16 * DH, DH, lane, k0, 1 << 30); };
  auto af_h1 = [&](int mt, int k0) { return afrag_b16(h1b + mt * 16 * DFF, DFF, lane, k0, 1 << 30); };

  float* xg = xglob + (size_t)b * ALPHA * DH;
  for (int i = tid; i < ALPHA * DH; i += 256) x[i] = xg[i];
  __syncthreads();

  for (int L = 0; L < NLl; ++L) {
    const bf16* WqT = wqt + (size_t)L * DH * DH;
    const bf16* WkT = wkt + (size_t)L * DH * DH;
    const bf16* WvT = wvt + (size_t)L * DH * DH;
    const bf16* WoT = wot + (size_t)L * DH * DH;
    const bf16* W1T = w1t + (size_t)L * DH * DFF;   // [512][128]
    const bf16* W2T = w2t + (size_t)L * DFF * DH;   // [128][512]

    // ---- Q / K / V projections: x(48x128) @ W(128x128) ----
    for (int t = 0; t < 3; ++t) {
      const bf16* WT = (t == 0) ? WqT : (t == 1) ? WkT : WvT;
      f32x8 acc[3] = {vz, vz, vz};
      gemm48_db(acc, WT, DH, 0, DH, bt0, bt1, tid, lane, w, af_x);
      if (t < 2) {
        bf16* dst = (t == 0) ? qb : kbq;
#pragma unroll
        for (int mt = 0; mt < 3; ++mt)
          cstore_b16(dst + mt * 16 * DH + w * 16, DH, lane, acc[mt]);
      } else {
#pragma unroll
        for (int mt = 0; mt < 3; ++mt)
          cstore_b16T(vbT + (w * 16) * ALPHA, ALPHA, mt * 16, lane, acc[mt]);
      }
    }
    __syncthreads();

    // ---- scores[h] = q_h(48x16) @ k_h^T / 4   (wave w = head, K padded 16->32)
    {
      f32x8 sc[3][3];
#pragma unroll
      for (int mt = 0; mt < 3; ++mt)
#pragma unroll
        for (int nt = 0; nt < 3; ++nt) sc[mt][nt] = vz;
      bf16x16 afs[3];
#pragma unroll
      for (int mt = 0; mt < 3; ++mt)
        afs[mt] = afrag_b16(qb + mt * 16 * DH + w * 16, DH, lane, 0, 16);
#pragma unroll
      for (int nt = 0; nt < 3; ++nt) {
        bf16x16 bfs = bfrag_rowT(kbq + nt * 16 * DH + w * 16, DH, lane, 0, 16);
#pragma unroll
        for (int mt = 0; mt < 3; ++mt)
          sc[mt][nt] = wmma_bf16(afs[mt], bfs, sc[mt][nt]);
      }
      float* sh = scores + w * (ALPHA * ALPHA);
#pragma unroll
      for (int mt = 0; mt < 3; ++mt)
#pragma unroll
        for (int nt = 0; nt < 3; ++nt)
          cstore_f32_scale(sh + mt * 16 * ALPHA + nt * 16, ALPHA, lane,
                           sc[mt][nt], 0.25f);  // 1/sqrt(hd=16)
    }
    __syncthreads();

    // ---- softmax over 48 keys, 8*48 rows ----
    for (int row = tid; row < Hn * ALPHA; row += 256) {
      float* p = scores + (row / ALPHA) * (ALPHA * ALPHA) + (row % ALPHA) * ALPHA;
      float mx = -3.4e38f;
#pragma unroll
      for (int j = 0; j < ALPHA; ++j) mx = fmaxf(mx, p[j]);
      float s = 0.f;
#pragma unroll
      for (int j = 0; j < ALPHA; ++j) { float e = __expf(p[j] - mx); p[j] = e; s += e; }
      float inv = 1.f / s;
#pragma unroll
      for (int j = 0; j < ALPHA; ++j) p[j] *= inv;
    }
    __syncthreads();

    // ---- o1_h = attn_h(48x48) @ v_h(48x16)   (wave w = head, K padded 48->64)
    {
      f32x8 oc[3] = {vz, vz, vz};
      for (int k0 = 0; k0 < 64; k0 += 32) {
        bf16x16 bfo = bfrag_rowT(vbT + (w * 16) * ALPHA, ALPHA, lane, k0, ALPHA);
#pragma unroll
        for (int mt = 0; mt < 3; ++mt) {
          bf16x16 af = afrag_f32(scores + w * (ALPHA * ALPHA) + mt * 16 * ALPHA,
                                 ALPHA, lane, k0, ALPHA);
          oc[mt] = wmma_bf16(af, bfo, oc[mt]);
        }
      }
#pragma unroll
      for (int mt = 0; mt < 3; ++mt)
        cstore_b16(o1b + mt * 16 * DH + w * 16, DH, lane, oc[mt]);
    }
    __syncthreads();

    // ---- x += o1 @ Wo ----
    {
      f32x8 acc[3] = {vz, vz, vz};
      gemm48_db(acc, WoT, DH, 0, DH, bt0, bt1, tid, lane, w, af_o1);
#pragma unroll
      for (int mt = 0; mt < 3; ++mt)
        cstore_f32_add(x + mt * 16 * DH + w * 16, DH, lane, acc[mt]);
    }
    __syncthreads();
    ln_rows(x, ln1g + L * DH, ln1b + L * DH, tid);
    __syncthreads();

    // ---- FFN1: h1 = relu(x @ W1 + b1), N=512 in 4 chunks of 128 ----
    for (int nc = 0; nc < 4; ++nc) {
      f32x8 acc[3] = {vz, vz, vz};
      gemm48_db(acc, W1T, DH, nc * 128, DH, bt0, bt1, tid, lane, w, af_x);
      int n  = nc * 128 + w * 16 + (lane & 15);
      int mb = (lane & 16) ? 8 : 0;
      float bv = fb1[L * DFF + n];
#pragma unroll
      for (int mt = 0; mt < 3; ++mt)
#pragma unroll
        for (int r = 0; r < 8; ++r) {
          float v = fmaxf(acc[mt][r] + bv, 0.f);
          h1b[(mt * 16 + mb + r) * DFF + n] = (bf16)v;
        }
    }
    __syncthreads();

    // ---- FFN2: x += h1 @ W2 + b2 ----
    {
      f32x8 acc[3] = {vz, vz, vz};
      gemm48_db(acc, W2T, DFF, 0, DFF, bt0, bt1, tid, lane, w, af_h1);
      int n  = w * 16 + (lane & 15);
      int mb = (lane & 16) ? 8 : 0;
      float bv = fb2[L * DH + n];
#pragma unroll
      for (int mt = 0; mt < 3; ++mt)
#pragma unroll
        for (int r = 0; r < 8; ++r)
          x[(mt * 16 + mb + r) * DH + n] += acc[mt][r] + bv;
    }
    __syncthreads();
    ln_rows(x, ln2g + L * DH, ln2b + L * DH, tid);
    __syncthreads();
  }

  // ---- classifier on last token (row 47) ----
  float* red = (float*)scr;
  if (tid < DH / 2) {
    const float* last = x + (ALPHA - 1) * DH;
    float a = cb1[tid];
#pragma unroll 8
    for (int f = 0; f < DH; ++f) a += last[f] * cw1[f * (DH / 2) + tid];
    a = fmaxf(a, 0.f);
    red[tid] = a * cw2[tid];
  }
  __syncthreads();
  if (tid == 0) {
    float s = cb2[0];
#pragma unroll
    for (int j = 0; j < DH / 2; ++j) s += red[j];
    out[b] = s;
  }
}

// ---------------------------------------------------------------------------
// Host launcher
// ---------------------------------------------------------------------------
extern "C" void kernel_launch(void* const* d_in, const int* in_sizes, int n_in,
                              void* d_out, int out_size, void* d_ws, size_t ws_size,
                              hipStream_t stream) {
  (void)in_sizes; (void)n_in; (void)out_size; (void)ws_size;

  const float* cls_emb     = (const float*)d_in[0];
  const float* note_times  = (const float*)d_in[1];
  const int*   note_counts = (const int*)  d_in[2];
  const float* proj_w      = (const float*)d_in[3];
  const float* proj_b      = (const float*)d_in[4];
  const float* te_w        = (const float*)d_in[5];
  const float* te_b        = (const float*)d_in[6];
  const float* wq          = (const float*)d_in[7];
  const float* wk          = (const float*)d_in[8];
  const float* tf_wq       = (const float*)d_in[9];
  const float* tf_wk       = (const float*)d_in[10];
  const float* tf_wv       = (const float*)d_in[11];
  const float* tf_wo       = (const float*)d_in[12];
  const float* tf_ln1_g    = (const float*)d_in[13];
  const float* tf_ln1_b    = (const float*)d_in[14];
  const float* tf_w1       = (const float*)d_in[15];
  const float* tf_b1       = (const float*)d_in[16];
  const float* tf_w2       = (const float*)d_in[17];
  const float* tf_b2       = (const float*)d_in[18];
  const float* tf_ln2_g    = (const float*)d_in[19];
  const float* tf_ln2_b    = (const float*)d_in[20];
  const float* cls_w1      = (const float*)d_in[21];
  const float* cls_b1      = (const float*)d_in[22];
  const float* cls_w2      = (const float*)d_in[23];
  const float* cls_b2      = (const float*)d_in[24];
  float* out = (float*)d_out;

  // workspace layout (bytes) -- all GEMM weights stored TRANSPOSED (W^T[N][K])
  unsigned char* ws = (unsigned char*)d_ws;
  bf16*  projwT = (bf16*)(ws + 0);               // [128][768] bf16 = 196608
  bf16*  wqt    = (bf16*)(ws + 196608);          // [3][128][128]   = 98304
  bf16*  wkt    = (bf16*)(ws + 294912);
  bf16*  wvt    = (bf16*)(ws + 393216);
  bf16*  wot    = (bf16*)(ws + 491520);
  bf16*  w1t    = (bf16*)(ws + 589824);          // [3][512][128]   = 393216
  bf16*  w2t    = (bf16*)(ws + 983040);          // [3][128][512]   = 393216
  float* qvec   = (float*)(ws + 1376256);        // 48*64 f32       = 12288
  bf16*  nembT  = (bf16*)(ws + 1388544);         // [256][128][64]  = 4194304
  float* xbuf   = (float*)(ws + 5582848);        // [256][48][128] f32 = 6291456
  // total = 11874304 bytes

  // K0: weight conversions (fp32 -> bf16, transposed)
  cvtT_f32_bf16<<<(98304  + 255) / 256, 256, 0, stream>>>(proj_w, projwT, D_T, DH, 98304);
  cvtT_f32_bf16<<<(49152  + 255) / 256, 256, 0, stream>>>(tf_wq, wqt, DH, DH, 49152);
  cvtT_f32_bf16<<<(49152  + 255) / 256, 256, 0, stream>>>(tf_wk, wkt, DH, DH, 49152);
  cvtT_f32_bf16<<<(49152  + 255) / 256, 256, 0, stream>>>(tf_wv, wvt, DH, DH, 49152);
  cvtT_f32_bf16<<<(49152  + 255) / 256, 256, 0, stream>>>(tf_wo, wot, DH, DH, 49152);
  cvtT_f32_bf16<<<(196608 + 255) / 256, 256, 0, stream>>>(tf_w1, w1t, DH, DFF, 196608);
  cvtT_f32_bf16<<<(196608 + 255) / 256, 256, 0, stream>>>(tf_w2, w2t, DFF, DH, 196608);

  // K0b: q-vector precompute
  qprep_kernel<<<1, 64, 0, stream>>>(te_w, te_b, wq, qvec);

  // K1: big projection GEMM (memory-bound; async double-buffered staging)
  proj_gemm_kernel<<<256, 256, 40960, stream>>>(cls_emb, projwT, proj_b, nembT);

  // K2: time-attention pooling -> x[b] (48x128)
  pool_kernel<<<Bsz, 256, 0, stream>>>(note_times, note_counts, te_w, te_b,
                                       wk, qvec, nembT, xbuf);

  // K3: fused 3-layer transformer + classifier (167936 B dynamic LDS / WG)
  tfm_kernel<<<Bsz, 256, 167936, stream>>>(xbuf, wqt, wkt, wvt, wot, w1t, w2t,
                                           tf_ln1_g, tf_ln1_b, tf_ln2_g, tf_ln2_b,
                                           tf_b1, tf_b2, cls_w1, cls_b1, cls_w2,
                                           cls_b2, out);
}
